// ABDRNN_33500744909433
// MI455X (gfx1250) — matrix operand order repaired and checked by reading, not compiled, not run on hardware
//
#include <hip/hip_runtime.h>
#include <hip/hip_bf16.h>

typedef __attribute__((ext_vector_type(16))) __bf16 v16bf;
typedef __attribute__((ext_vector_type(8)))  float  v8f;

union Frag {
    uint4 q[2];
    v16bf v;
};

union Pack16 {
    unsigned short s[16];
    uint4 q[2];
};

__device__ __forceinline__ unsigned short f32_to_bf16_rne(float f) {
    union { float f; unsigned int u; } v; v.f = f;
    unsigned int u = v.u;
    unsigned int r = u + 0x7FFFu + ((u >> 16) & 1u);  // round-to-nearest-even
    return (unsigned short)(r >> 16);
}

// C[Mrows,N] = A[Mrows,K] * W[N,K]^T + bias[N]
// Block tile 128x128, K-step 32, 256 threads = 8 waves, each wave 32x64.
__global__ __launch_bounds__(256)
void abdrnn_gemm_bias(const float* __restrict__ A,
                      const float* __restrict__ W,
                      const float* __restrict__ bias,
                      float* __restrict__ C,
                      int Mrows, int N, int K)
{
    constexpr int BK     = 32;
    constexpr int LDSTR  = 40;   // row stride in bf16 elems: 80 bytes, 16B aligned, conflict-free
    __shared__ __align__(16) unsigned short As[128 * LDSTR];
    __shared__ __align__(16) unsigned short Bs[128 * LDSTR];

    const int tid      = threadIdx.x;
    const int rowBase  = blockIdx.y * 128;   // along T*M
    const int colBase  = blockIdx.x * 128;   // along H

    // loader mapping: each thread moves 16 consecutive floats (one half-row of a tile row)
    const int ldRow = tid >> 1;          // 0..127
    const int ldCol = (tid & 1) * 16;    // 0 or 16

    // wave mapping: wn in 0..1 (64-col groups), wm in 0..3 (32-row groups)
    const int wave = tid >> 5;
    const int lane = tid & 31;
    const int lo16 = lane & 15;
    const int hi   = lane >> 4;          // 0 or 1
    const int wn   = wave >> 2;          // 0..1
    const int wm   = wave & 3;           // 0..3
    const int rB   = wm * 32;            // wave row offset in block tile
    const int cB   = wn * 64;            // wave col offset in block tile

    v8f acc[2][4];
    #pragma unroll
    for (int i = 0; i < 2; ++i)
        #pragma unroll
        for (int j = 0; j < 4; ++j)
            acc[i][j] = (v8f){0.f, 0.f, 0.f, 0.f, 0.f, 0.f, 0.f, 0.f};

    for (int kb = 0; kb < K; kb += BK) {
        // ---- stage fp32 -> bf16 into LDS (A tile and B tile, both K-contiguous) ----
        {
            const float4* ga = (const float4*)(A + (size_t)(rowBase + ldRow) * K + kb + ldCol);
            const float4* gb = (const float4*)(W + (size_t)(colBase + ldRow) * K + kb + ldCol);
            float4 fa0 = ga[0], fa1 = ga[1], fa2 = ga[2], fa3 = ga[3];
            float4 fb0 = gb[0], fb1 = gb[1], fb2 = gb[2], fb3 = gb[3];
            Pack16 pa, pb;
            const float av[16] = {fa0.x,fa0.y,fa0.z,fa0.w, fa1.x,fa1.y,fa1.z,fa1.w,
                                  fa2.x,fa2.y,fa2.z,fa2.w, fa3.x,fa3.y,fa3.z,fa3.w};
            const float bv[16] = {fb0.x,fb0.y,fb0.z,fb0.w, fb1.x,fb1.y,fb1.z,fb1.w,
                                  fb2.x,fb2.y,fb2.z,fb2.w, fb3.x,fb3.y,fb3.z,fb3.w};
            #pragma unroll
            for (int j = 0; j < 16; ++j) { pa.s[j] = f32_to_bf16_rne(av[j]); pb.s[j] = f32_to_bf16_rne(bv[j]); }
            uint4* da = (uint4*)(As + ldRow * LDSTR + ldCol);
            uint4* db = (uint4*)(Bs + ldRow * LDSTR + ldCol);
            da[0] = pa.q[0]; da[1] = pa.q[1];
            db[0] = pb.q[0]; db[1] = pb.q[1];
        }
        __syncthreads();

        // ---- B fragments for this wave: 4 subtiles of 16 cols ----
        Frag fb[4];
        #pragma unroll
        for (int ns = 0; ns < 4; ++ns) {
            const unsigned short* bp = Bs + (cB + ns * 16 + lo16) * LDSTR + hi * 16;
            fb[ns].q[0] = *(const uint4*)(bp);
            fb[ns].q[1] = *(const uint4*)(bp + 8);
        }

        // ---- A fragments + WMMA ----
        #pragma unroll
        for (int ms = 0; ms < 2; ++ms) {
            Frag fa;
            const unsigned short* ap = As + (rB + ms * 16 + lo16) * LDSTR;
            fa.q[0] = *(const uint4*)(ap + hi * 16);
            fa.q[1] = *(const uint4*)(ap + hi * 16 + 8);
            #pragma unroll
            for (int ns = 0; ns < 4; ++ns) {
                acc[ms][ns] = __builtin_amdgcn_wmma_f32_16x16x32_bf16(
                    /*neg_a=*/false, fa.v, /*neg_b=*/false, fb[ns].v,
                    /*c_mod=*/(short)0, acc[ms][ns],
                    /*reuse_a=*/false, /*reuse_b=*/false);
            }
        }
        __syncthreads();
    }

    // ---- epilogue: bias add + store (C/D layout: lane = col, elem r = row) ----
    #pragma unroll
    for (int ms = 0; ms < 2; ++ms) {
        #pragma unroll
        for (int ns = 0; ns < 4; ++ns) {
            const int col  = colBase + cB + ns * 16 + lo16;
            const float bv = bias[col];
            const int row0 = rowBase + rB + ms * 16 + hi * 8;
            float* cp = C + (size_t)row0 * N + col;
            #pragma unroll
            for (int r = 0; r < 8; ++r) {
                *cp = acc[ms][ns][r] + bv;
                cp += N;
            }
        }
    }
}

// In-place recurrence over T: wx[t] -> all_h[t]; 4 independent chains per thread.
__global__ __launch_bounds__(256)
void abdrnn_scan(float4* __restrict__ wx,          // [T, MH4] in-place
                 const float4* __restrict__ h0,    // [MH4]
                 const float* __restrict__ rptr,   // scalar rgate
                 float4* __restrict__ hn_out,      // [MH4]
                 int T, int MH4)
{
    const int idx = blockIdx.x * blockDim.x + threadIdx.x;
    if (idx >= MH4) return;
    const float r = *rptr;
    const float q = 1.0f - r;
    float4 h = h0[idx];
    float4* p = wx + idx;
    for (int t = 0; t < T; ++t) {
        float4 w = *p;
        h.x = fmaxf(0.0f, fmaf(r, h.x, q * w.x));
        h.y = fmaxf(0.0f, fmaf(r, h.y, q * w.y));
        h.z = fmaxf(0.0f, fmaf(r, h.z, q * w.z));
        h.w = fmaxf(0.0f, fmaf(r, h.w, q * w.w));
        *p = h;
        p += MH4;
    }
    hn_out[idx] = h;
}

extern "C" void kernel_launch(void* const* d_in, const int* in_sizes, int n_in,
                              void* d_out, int out_size, void* d_ws, size_t ws_size,
                              hipStream_t stream) {
    (void)in_sizes; (void)n_in; (void)out_size; (void)ws_size;
    constexpr int T = 512, M = 64, I = 1024, H = 1024;
    constexpr int Mrows = T * M;        // 32768
    constexpr int MH    = M * H;        // 65536
    constexpr int MH4   = MH / 4;       // 16384

    const float* x   = (const float*)d_in[0];
    const float* h_0 = (const float*)d_in[1];
    const float* W0  = (const float*)d_in[2];
    const float* b0  = (const float*)d_in[3];
    const float* r0  = (const float*)d_in[4];
    const float* W1  = (const float*)d_in[5];
    const float* b1  = (const float*)d_in[6];
    const float* r1  = (const float*)d_in[7];

    float* out    = (float*)d_out;           // [T,M,H] all_h1, then [2,M,H] h_n
    float* hn     = out + (size_t)T * MH;    // h_n base
    float* wx0    = (float*)d_ws;            // [T,M,H] scratch (becomes all_h0)

    dim3 gGemm(H / 128, Mrows / 128);        // (8, 256)
    dim3 bGemm(256);
    dim3 gScan((MH4 + 255) / 256);           // 64 blocks
    dim3 bScan(256);

    // Layer 0: wx0 = x @ W0^T + b0   (K = I)
    abdrnn_gemm_bias<<<gGemm, bGemm, 0, stream>>>(x, W0, b0, wx0, Mrows, H, I);
    // Layer 0 scan: ws in-place -> all_h0 ; hn0 -> d_out tail
    abdrnn_scan<<<gScan, bScan, 0, stream>>>((float4*)wx0, (const float4*)(h_0),
                                             r0, (float4*)hn, T, MH4);
    // Layer 1: wx1 = all_h0 @ W1^T + b1 -> d_out (K = H)
    abdrnn_gemm_bias<<<gGemm, bGemm, 0, stream>>>(wx0, W1, b1, out, Mrows, H, H);
    // Layer 1 scan: d_out in-place -> all_h1 ; hn1 -> d_out tail + MH
    abdrnn_scan<<<gScan, bScan, 0, stream>>>((float4*)out, (const float4*)(h_0 + MH),
                                             r1, (float4*)(hn + MH), T, MH4);
}
